// Actor_51728586113025
// MI455X (gfx1250) — matrix operand order repaired and checked by reading, not compiled, not run on hardware
//
#include <hip/hip_runtime.h>
#include <hip/hip_bf16.h>

typedef __attribute__((ext_vector_type(16))) _Float16 v16h;
typedef __attribute__((ext_vector_type(8)))  float    v8f;

constexpr int BB = 8;
constexpr int NN = 1024;
constexpr int FF = 128;
constexpr int EE = 64;

// ---------------------------------------------------------------------------
// Stage 1: per-row MLP. kv = ssp(h@W1+b1)@W2+b2, masked; write k,v as f16.
// K is pre-scaled by 1/sqrt(E)=0.125 so the hot loop needs no extra multiply.
// Grid: (B*N/16) blocks x 256 threads; each block does 16 rows.
// ---------------------------------------------------------------------------
__global__ __launch_bounds__(256) void actor_mlp_kernel(
    const float* __restrict__ h, const float* __restrict__ mask,
    const float* __restrict__ W1, const float* __restrict__ bias1,
    const float* __restrict__ W2, const float* __restrict__ bias2,
    _Float16* __restrict__ Kh, _Float16* __restrict__ Vh)
{
  __shared__ float sh[16 * FF];    // 8 KB h tile
  __shared__ float sact[16 * EE];  // 4 KB activations
  const int b    = blockIdx.x >> 6;          // 64 row-tiles per batch
  const int row0 = (blockIdx.x & 63) << 4;
  const int tid  = threadIdx.x;

  const float* hbase = h + (size_t)(b * NN + row0) * FF;
  for (int idx = tid; idx < 16 * FF; idx += 256) sh[idx] = hbase[idx];
  __syncthreads();

  // layer 1 + shifted softplus: 16 rows x 64 mid
  for (int idx = tid; idx < 16 * EE; idx += 256) {
    const int r = idx >> 6, e = idx & 63;
    float s = bias1[e];
    const float* hr = &sh[r * FF];
#pragma unroll 8
    for (int f = 0; f < FF; ++f) s = fmaf(hr[f], W1[f * EE + e], s);
    // numerically stable softplus
    const float sp = (s > 0.f) ? (s + log1pf(expf(-s))) : log1pf(expf(s));
    sact[idx] = sp - 0.6931471805599453f;
  }
  __syncthreads();

  // layer 2: only the first 128 of 129 outputs are ever used (k | v)
  for (int idx = tid; idx < 16 * 128; idx += 256) {
    const int r = idx >> 7, o = idx & 127;
    float s = bias2[o];
    const float* ar = &sact[r * EE];
#pragma unroll 8
    for (int e = 0; e < EE; ++e) s = fmaf(ar[e], W2[e * 129 + o], s);
    s *= mask[b * NN + row0 + r];
    const size_t gi = (size_t)(b * NN + row0 + r) * EE;
    if (o < EE) Kh[gi + o]        = (_Float16)(s * 0.125f);  // fold 1/sqrt(E)
    else        Vh[gi + (o - EE)] = (_Float16)s;
  }
}

// ---------------------------------------------------------------------------
// Stage 2: fused  S = K·V^T/8  and  a_i = sum_j S_ij * unit(p_i - p_j).
// Each wave owns one 16-row i-tile; streams 64 j-tiles of 16 cols.
// Two v_wmma_f32_16x16x32_f16 per j-tile (K=64 contraction); one v_rsq_f32
// per pair-distance (diagonal stays exactly zero via the d2 clamp, since
// dx=dy=dz=0 multiplies the finite rsq result). Positions live in LDS.
// Grid: B*(N/32) blocks x 64 threads (2 waves).
// ---------------------------------------------------------------------------
__global__ __launch_bounds__(64) void actor_attn_kernel(
    const float* __restrict__ pos, const float* __restrict__ mask,
    const _Float16* __restrict__ Kh, const _Float16* __restrict__ Vh,
    float* __restrict__ out)
{
  __shared__ float sx[NN], sy[NN], sz[NN];   // 12 KB positions for this batch
  __shared__ float sred[64];
  __shared__ float smsum;

  const int b    = blockIdx.x >> 5;   // 32 blocks per batch
  const int pair = blockIdx.x & 31;
  const int tid  = threadIdx.x;
  const int wave = tid >> 5;
  const int lane = tid & 31;
  const int hf   = lane >> 4;         // wave32 half (WMMA layout)
  const int ln   = lane & 15;
  const int i0   = ((pair << 1) + wave) << 4;

  // load positions + mask sum (= atoms_mask.sum(-1))
  float mp = 0.f;
  for (int j = tid; j < NN; j += 64) {
    const float* p = pos + (size_t)(b * NN + j) * 3;
    sx[j] = p[0]; sy[j] = p[1]; sz[j] = p[2];
    mp += mask[b * NN + j];
  }
  sred[tid] = mp;
  __syncthreads();
  if (tid == 0) { float t = 0.f; for (int k = 0; k < 64; ++k) t += sred[k]; smsum = t; }
  __syncthreads();
  const float inv_msum = 1.0f / smsum;

  // A tile (16x64 of K, f16): ISA layout = two 16B contiguous loads per K=32 chunk
  const _Float16* arow = Kh + (size_t)(b * NN + i0 + ln) * EE;
  v16h a0, a1;
  ((uint4*)&a0)[0] = *(const uint4*)(arow +      8 * hf);
  ((uint4*)&a0)[1] = *(const uint4*)(arow + 16 + 8 * hf);
  ((uint4*)&a1)[0] = *(const uint4*)(arow + 32 + 8 * hf);
  ((uint4*)&a1)[1] = *(const uint4*)(arow + 48 + 8 * hf);

  // C/D layout: lane ln holds N=ln, VGPR r holds M = r + 8*hf
  float pix[8], piy[8], piz[8];
  float ax[8] = {0}, ay[8] = {0}, az[8] = {0};
#pragma unroll
  for (int r = 0; r < 8; ++r) {
    const int i = i0 + r + 8 * hf;
    pix[r] = sx[i]; piy[r] = sy[i]; piz[r] = sz[i];
  }

  for (int j0 = 0; j0 < NN; j0 += 16) {
    const _Float16* vrow = Vh + (size_t)(b * NN + j0 + ln) * EE;
    if (j0 + 16 < NN) __builtin_prefetch(vrow + 16 * EE, 0, 0);  // global_prefetch_b8

    // B tile (32x16 f16): lane ln = column N, halves = K 8*hf..+7 and 16+8*hf..+7 (same as A)
    v16h vb0, vb1;
    ((uint4*)&vb0)[0] = *(const uint4*)(vrow +      8 * hf);
    ((uint4*)&vb0)[1] = *(const uint4*)(vrow + 16 + 8 * hf);
    ((uint4*)&vb1)[0] = *(const uint4*)(vrow + 32 + 8 * hf);
    ((uint4*)&vb1)[1] = *(const uint4*)(vrow + 48 + 8 * hf);

    v8f acc = {};
    acc = __builtin_amdgcn_wmma_f32_16x16x32_f16(false, a0, false, vb0,
                                                 (short)0, acc, false, false);
    acc = __builtin_amdgcn_wmma_f32_16x16x32_f16(false, a1, false, vb1,
                                                 (short)0, acc, false, false);

    const int j = j0 + ln;
    const float qx = sx[j], qy = sy[j], qz = sz[j];
#pragma unroll
    for (int r = 0; r < 8; ++r) {
      const float s  = acc[r];                         // 1/sqrt(E) pre-folded
      const float dx = pix[r] - qx;
      const float dy = piy[r] - qy;
      const float dz = piz[r] - qz;
      const float d2 = fmaf(dx, dx, fmaf(dy, dy, dz * dz));
      // single v_rsq_f32; clamp keeps diagonal finite (then *0 = 0 exactly)
      const float w  = s * __builtin_amdgcn_rsqf(fmaxf(d2, 1e-24f));
      ax[r] = fmaf(w, dx, ax[r]);
      ay[r] = fmaf(w, dy, ay[r]);
      az[r] = fmaf(w, dz, az[r]);
    }
  }

  // reduce over the 16 N-lanes of each half (xor masks stay within the half)
#pragma unroll
  for (int m = 1; m < 16; m <<= 1) {
#pragma unroll
    for (int r = 0; r < 8; ++r) {
      ax[r] += __shfl_xor(ax[r], m, 32);
      ay[r] += __shfl_xor(ay[r], m, 32);
      az[r] += __shfl_xor(az[r], m, 32);
    }
  }

  if (ln == 0) {
#pragma unroll
    for (int r = 0; r < 8; ++r) {
      const int   i  = i0 + r + 8 * hf;
      const float mk = mask[b * NN + i];     // ACTION_SCALE == 1
      float* o = out + (size_t)(b * NN + i) * 3;
      o[0] = tanhf(ax[r] * inv_msum) * mk;
      o[1] = tanhf(ay[r] * inv_msum) * mk;
      o[2] = tanhf(az[r] * inv_msum) * mk;
    }
  }
}

extern "C" void kernel_launch(void* const* d_in, const int* in_sizes, int n_in,
                              void* d_out, int out_size, void* d_ws, size_t ws_size,
                              hipStream_t stream) {
  const float* positions  = (const float*)d_in[0];
  const float* atoms_mask = (const float*)d_in[1];
  const float* h          = (const float*)d_in[2];
  const float* W1         = (const float*)d_in[3];
  const float* b1         = (const float*)d_in[4];
  const float* W2         = (const float*)d_in[5];
  const float* b2         = (const float*)d_in[6];
  float* out = (float*)d_out;

  // workspace: K (1 MB f16, pre-scaled by 1/sqrt(E)) | V (1 MB f16)
  _Float16* Kh = (_Float16*)d_ws;
  _Float16* Vh = Kh + (size_t)BB * NN * EE;

  actor_mlp_kernel<<<BB * NN / 16, 256, 0, stream>>>(h, atoms_mask, W1, b1, W2, b2, Kh, Vh);
  actor_attn_kernel<<<BB * (NN / 32), 64, 0, stream>>>(positions, atoms_mask, Kh, Vh, out);
}
